// BiLSTM_CRF_49297634623761
// MI455X (gfx1250) — matrix operand order repaired and checked by reading, not compiled, not run on hardware
//
#include <hip/hip_runtime.h>
#include <hip/hip_bf16.h>
#include <math.h>

typedef __attribute__((ext_vector_type(16))) __bf16 v16bf;
typedef __attribute__((ext_vector_type(8)))  float  v8f;
typedef __attribute__((ext_vector_type(4)))  unsigned int u32x4;
typedef __attribute__((ext_vector_type(8)))  int i32x8;
typedef __attribute__((ext_vector_type(4)))  int i32x4;

#define SLEN 2048
#define EDIM 512
#define HDIM 512      // per-direction hidden
#define G4   2048     // 4*HDIM gate rows
#define HCAT 1024     // concat hidden
#define NT   12
#define NEGV -1.0e6f

#if defined(__HIP_DEVICE_COMPILE__) && __has_builtin(__builtin_amdgcn_tensor_load_to_lds)
#define HAVE_TDM 1
#else
#define HAVE_TDM 0
#endif

// ---------- helpers ----------
__device__ inline unsigned short f2bf(float f){
  unsigned u = __float_as_uint(f);
  u += 0x7FFFu + ((u >> 16) & 1u);          // round-to-nearest-even
  return (unsigned short)(u >> 16);
}

// ---------- prep kernels ----------
__global__ void k_gather_x(const int* __restrict__ sent, const float* __restrict__ emb,
                           unsigned short* __restrict__ xbf){
  int i = blockIdx.x * blockDim.x + threadIdx.x;
  if (i >= SLEN * EDIM) return;
  int t = i >> 9;            // /512
  int e = i & 511;
  xbf[i] = f2bf(emb[(size_t)sent[t] * EDIM + e]);
}

__global__ void k_cvt(const float* __restrict__ src, unsigned short* __restrict__ dst, int n){
  int i = blockIdx.x * blockDim.x + threadIdx.x;
  if (i < n) dst[i] = f2bf(src[i]);
}

__global__ void k_zero(int* __restrict__ p, int n){
  int i = blockIdx.x * blockDim.x + threadIdx.x;
  if (i < n) p[i] = 0;
}

// ---------- WMMA GEMM: xproj[dir][t][r] = sum_e x[t][e]*wih[r][e] + b[r] ----------
struct BF16x16 { uint4 lo, hi; };

// A-matrix 16x32 bf16 fragment per ISA layout:
// lane l (row M=l%16): e=0..7 -> K = base8+e ; e=8..15 -> K = base8+e+8 ; base8 = (l<16?0:8)
__device__ inline v16bf load_a_frag(const unsigned short* __restrict__ X,
                                    int rowBase, int kBase, int lane){
  int row   = rowBase + (lane & 15);
  int base8 = (lane < 16) ? 0 : 8;
  const unsigned short* p = X + (size_t)row * EDIM + kBase + base8;
  BF16x16 t;
  t.lo = *(const uint4*)(p);        // K = base8 + 0..7
  t.hi = *(const uint4*)(p + 16);   // K = base8 + 16..23
  return __builtin_bit_cast(v16bf, t);
}

// B-matrix 32x16 bf16 fragment: lane l (col N=l%16): e=0..7 -> K = 8*(l/16)+e ; e=8..15 -> K = 8*(l/16)+e+8
// value = wih[N][K] (row-major wih => two runs of 8 contiguous bf16)
__device__ inline v16bf load_b_frag(const unsigned short* __restrict__ W,
                                    int colBase, int kBase, int lane){
  int n  = colBase + (lane & 15);
  int kb = kBase + ((lane >> 4) << 3);
  const unsigned short* p = W + (size_t)n * EDIM + kb;
  BF16x16 t;
  t.lo = *(const uint4*)(p);
  t.hi = *(const uint4*)(p + 16);
  return __builtin_bit_cast(v16bf, t);
}

__global__ __launch_bounds__(256)
void k_xproj(const unsigned short* __restrict__ xbf, const unsigned short* __restrict__ wihbf,
             const float* __restrict__ b_f, const float* __restrict__ b_b,
             float* __restrict__ xproj){
  const int dir = blockIdx.z;
  const unsigned short* W = wihbf + (size_t)dir * G4 * EDIM;
  const float* bias = dir ? b_b : b_f;
  float* outp = xproj + (size_t)dir * SLEN * G4;

  const int tid   = threadIdx.x;
  const int lane  = tid & 31;
  const int wave  = tid >> 5;       // 0..7
  const int waveM = wave & 1;       // 2 row groups
  const int waveN = wave >> 1;      // 4 col groups
  const int rowBase = blockIdx.y * 64  + waveM * 32;   // M (sequence t)
  const int colBase = blockIdx.x * 256 + waveN * 64;   // N (gate row)

  v8f acc[4][2];
  #pragma unroll
  for (int c = 0; c < 4; ++c)
    #pragma unroll
    for (int r = 0; r < 2; ++r)
      acc[c][r] = (v8f){0.f,0.f,0.f,0.f,0.f,0.f,0.f,0.f};

  for (int k = 0; k < EDIM; k += 32){
    v16bf a0 = load_a_frag(xbf, rowBase,      k, lane);
    v16bf a1 = load_a_frag(xbf, rowBase + 16, k, lane);
    v16bf bfr[4];
    #pragma unroll
    for (int c = 0; c < 4; ++c)
      bfr[c] = load_b_frag(W, colBase + c*16, k, lane);   // issue all loads first
    #pragma unroll
    for (int c = 0; c < 4; ++c){
      acc[c][0] = __builtin_amdgcn_wmma_f32_16x16x32_bf16(false, a0, false, bfr[c],
                      (short)0, acc[c][0], false, false);
      acc[c][1] = __builtin_amdgcn_wmma_f32_16x16x32_bf16(false, a1, false, bfr[c],
                      (short)0, acc[c][1], false, false);
    }
  }

  // C/D layout: VGPR r -> M = 8*(l/16)+r ; N = l%16
  const int laneN  = lane & 15;
  const int laneHi = lane >> 4;
  #pragma unroll
  for (int c = 0; c < 4; ++c){
    int col = colBase + c*16 + laneN;
    float bc = bias[col];
    #pragma unroll
    for (int rt = 0; rt < 2; ++rt){
      #pragma unroll
      for (int r = 0; r < 8; ++r){
        int row = rowBase + rt*16 + laneHi*8 + r;
        outp[(size_t)row * G4 + col] = acc[c][rt][r] + bc;
      }
    }
  }
}

// ---------- recurrent LSTM: 32 workgroups per direction, Whh slice resident in LDS ----------
__global__ __launch_bounds__(256)
void k_lstm(const float* __restrict__ whh_f, const float* __restrict__ whh_b,
            const float* __restrict__ h0, const float* __restrict__ c0,
            const float* __restrict__ xproj, float* __restrict__ hbuf,
            float* __restrict__ hs, int* __restrict__ bar){
  const int blk  = blockIdx.x;       // 0..63
  const int dir  = blk >> 5;
  const int part = blk & 31;
  const int u0   = part * 16;        // 16 hidden units per WG
  const float* whh    = dir ? whh_b : whh_f;
  const float* xprojD = xproj + (size_t)dir * SLEN * G4;
  const int tid = threadIdx.x;

  __shared__ float Wsh[64][EDIM];    // 128 KB  (CDNA5: up to 320 KB LDS/WGP)
  __shared__ float hsh[HDIM];
  __shared__ float gv[64];
  __shared__ float csh[16];

#if HAVE_TDM
  // Tensor Data Mover: DMA the 4 gate tiles (16 rows x 512 f32, row stride 512)
  // straight into LDS; issued by wave 0, tracked by TENSORcnt.
  if (tid < 32){
    #pragma unroll
    for (int g = 0; g < 4; ++g){
      unsigned long long gaddr =
          (unsigned long long)(uintptr_t)(whh + ((size_t)g * HDIM + u0) * EDIM);
      unsigned laddr = (unsigned)(uintptr_t)(&Wsh[g * 16][0]);  // LDS offset = addr[31:0]
      u32x4 d0;
      d0.x = 1u;                                         // count=1, user mode
      d0.y = laddr;                                      // lds_addr
      d0.z = (unsigned)(gaddr & 0xFFFFFFFFull);          // global_addr[31:0]
      d0.w = (unsigned)((gaddr >> 32) & 0x01FFFFFFull) | (2u << 30);  // addr[56:32]|type=2
      i32x8 d1;
      d1[0] = (int)(2u << 16);                           // data_size=4B, wg_mask=0
      d1[1] = (int)((EDIM & 0xFFFF) << 16);              // tensor_dim0 lo16
      d1[2] = (int)((EDIM >> 16) | ((G4 & 0xFFFF) << 16)); // dim0 hi16 | tensor_dim1 lo16
      d1[3] = (int)((G4 >> 16) | (EDIM << 16));          // dim1 hi16 | tile_dim0=512
      d1[4] = 16;                                        // tile_dim1=16, tile_dim2=0
      d1[5] = EDIM;                                      // tensor_dim0_stride lo32
      d1[6] = 0;
      d1[7] = 0;
      i32x4 z4 = {0, 0, 0, 0};
#if defined(__clang_major__) && (__clang_major__ >= 23)
      i32x8 z8 = {0, 0, 0, 0, 0, 0, 0, 0};
      __builtin_amdgcn_tensor_load_to_lds(d0, d1, z4, z4, z8, 0);
#else
      __builtin_amdgcn_tensor_load_to_lds(d0, d1, z4, z4, 0);
#endif
    }
    __builtin_amdgcn_s_wait_tensorcnt(0);
  }
#else
  // fallback: cooperative float4 staging
  for (int i = tid; i < 64 * (EDIM/4); i += 256){
    int lr = i >> 7;
    int ch = i & 127;
    int grow = (lr >> 4) * HDIM + u0 + (lr & 15);
    float4 v = *(const float4*)(whh + (size_t)grow * EDIM + ch*4);
    *(float4*)(&Wsh[lr][ch*4]) = v;
  }
#endif
  if (tid < 16) csh[tid] = c0[dir*HDIM + u0 + tid];

  const int lr   = tid >> 2;          // row 0..63, 4 lanes per row
  const int q    = tid & 3;           // quarter of K
  const int grow = (lr >> 4) * HDIM + u0 + (lr & 15);
  int* mybar = bar + dir * SLEN;

  for (int s = 0; s < SLEN; ++s){
    const int t = dir ? (SLEN - 1 - s) : s;
    const float* hsrc = (s == 0) ? (h0 + dir*HDIM) : (hbuf + dir*HDIM);
    __syncthreads();                  // protect hsh from previous iteration's readers
    hsh[tid]       = hsrc[tid];
    hsh[tid + 256] = hsrc[tid + 256];
    __syncthreads();

    // gate = Whh_row . h  (4-lane split of K)
    float p = 0.f;
    const float* wrow = &Wsh[lr][q * 128];
    const float* hseg = &hsh[q * 128];
    #pragma unroll 8
    for (int e = 0; e < 128; ++e) p = fmaf(wrow[e], hseg[e], p);
    p += __shfl_xor(p, 1, 32);
    p += __shfl_xor(p, 2, 32);
    if (q == 0) gv[lr] = p + xprojD[(size_t)t * G4 + grow];
    if (q == 0 && s + 1 < SLEN){      // gfx1250 global_prefetch_b8 for next step
      int tn = dir ? (SLEN - 2 - s) : (s + 1);
      __builtin_prefetch(&xprojD[(size_t)tn * G4 + grow], 0, 0);
    }
    __syncthreads();

    if (tid < 16){                    // i,f,g,o order (rows j, 512+j, 1024+j, 1536+j)
      float gi = gv[tid], gf = gv[16+tid], gg = gv[32+tid], go = gv[48+tid];
      float i_ = 1.f / (1.f + __expf(-gi));
      float f_ = 1.f / (1.f + __expf(-gf));
      float o_ = 1.f / (1.f + __expf(-go));
      float c  = f_ * csh[tid] + i_ * tanhf(gg);
      csh[tid] = c;
      float h  = o_ * tanhf(c);
      hbuf[dir*HDIM + u0 + tid] = h;
      hs[(size_t)t * HCAT + dir*HDIM + u0 + tid] = h;
    }
    __threadfence();
    __syncthreads();
    if (tid == 0){                    // device-scope barrier across the 32 WGs of this dir
      atomicAdd(&mybar[s], 1);
      while (__hip_atomic_load(&mybar[s], __ATOMIC_ACQUIRE, __HIP_MEMORY_SCOPE_AGENT) < 32) { }
    }
    __syncthreads();
    __threadfence();
  }
}

// ---------- feats = [hf;hb] @ w_out.T + b_out : one wave per tag ----------
__global__ __launch_bounds__(384)
void k_feats(const float* __restrict__ hs, const float* __restrict__ w_out,
             const float* __restrict__ b_out, float* __restrict__ feats){
  const int t    = blockIdx.x;
  const int tag  = threadIdx.x >> 5;   // 0..11
  const int lane = threadIdx.x & 31;
  const float* hrow = hs + (size_t)t * HCAT;
  const float* wrow = w_out + (size_t)tag * HCAT;
  float p = 0.f;
  for (int i = lane; i < HCAT; i += 32) p = fmaf(wrow[i], hrow[i], p);
  #pragma unroll
  for (int o = 16; o > 0; o >>= 1) p += __shfl_down(p, o, 32);
  if (lane == 0) feats[t * NT + tag] = p + b_out[tag];
}

// ---------- CRF: forward scan + gold score ----------
__global__ __launch_bounds__(256)
void k_crf(const float* __restrict__ feats, const int* __restrict__ gold,
           const float* __restrict__ trans, float* __restrict__ out){
  __shared__ float tr[NT * NT];
  __shared__ float prev[NT], nxt[NT];
  __shared__ float red[256];
  __shared__ float alpha_sh;
  const int tid = threadIdx.x;
  if (tid < NT * NT) tr[tid] = trans[tid];
  if (tid < NT) prev[tid] = (tid == 0) ? 0.f : NEGV;   // START = 0
  __syncthreads();

  for (int t = 0; t < SLEN; ++t){
    if (tid < NT){
      float m = -3.4e38f;
      #pragma unroll
      for (int j = 0; j < NT; ++j) m = fmaxf(m, prev[j] + tr[tid*NT + j]);
      float sum = 0.f;
      #pragma unroll
      for (int j = 0; j < NT; ++j) sum += __expf(prev[j] + tr[tid*NT + j] - m);
      nxt[tid] = m + __logf(sum) + feats[t * NT + tid];
    }
    __syncthreads();
    if (tid < NT) prev[tid] = nxt[tid];
    __syncthreads();
  }
  if (tid == 0){
    float m = -3.4e38f;
    for (int i = 0; i < NT; ++i) m = fmaxf(m, prev[i] + tr[1*NT + i]);  // END = 1
    float sum = 0.f;
    for (int i = 0; i < NT; ++i) sum += __expf(prev[i] + tr[1*NT + i] - m);
    alpha_sh = m + __logf(sum);
  }
  // gold score (parallel over t)
  float g = 0.f;
  for (int t = tid; t < SLEN; t += 256){
    int cur = gold[t];
    int pv  = (t == 0) ? 0 : gold[t-1];
    g += tr[cur*NT + pv] + feats[t*NT + cur];
  }
  red[tid] = g;
  __syncthreads();
  for (int o = 128; o > 0; o >>= 1){
    if (tid < o) red[tid] += red[tid + o];
    __syncthreads();
  }
  if (tid == 0){
    float goldScore = red[0] + tr[1*NT + gold[SLEN-1]];
    out[0] = alpha_sh - goldScore;
  }
}

// ---------- launcher ----------
extern "C" void kernel_launch(void* const* d_in, const int* in_sizes, int n_in,
                              void* d_out, int out_size, void* d_ws, size_t ws_size,
                              hipStream_t stream){
  const int*   sent  = (const int*)  d_in[0];
  const int*   gold  = (const int*)  d_in[1];
  const float* emb   = (const float*)d_in[2];
  const float* wih_f = (const float*)d_in[3];
  const float* whh_f = (const float*)d_in[4];
  const float* b_f   = (const float*)d_in[5];
  const float* wih_b = (const float*)d_in[6];
  const float* whh_b = (const float*)d_in[7];
  const float* b_b   = (const float*)d_in[8];
  const float* w_out = (const float*)d_in[9];
  const float* b_out = (const float*)d_in[10];
  const float* trans = (const float*)d_in[11];
  const float* h0    = (const float*)d_in[12];
  const float* c0    = (const float*)d_in[13];
  float* out = (float*)d_out;

  char* ws = (char*)d_ws;
  size_t off = 0;
  auto wsalloc = [&](size_t bytes) -> char* {
    char* p = ws + off;
    off = (off + bytes + 255) & ~(size_t)255;
    return p;
  };
  float*          xproj = (float*)         wsalloc(sizeof(float) * 2 * SLEN * G4);   // 32 MB
  float*          hs    = (float*)         wsalloc(sizeof(float) * SLEN * HCAT);     // 8 MB
  float*          hbuf  = (float*)         wsalloc(sizeof(float) * 2 * HDIM);
  int*            bar   = (int*)           wsalloc(sizeof(int)   * 2 * SLEN);
  float*          feats = (float*)         wsalloc(sizeof(float) * SLEN * NT);
  unsigned short* xbf   = (unsigned short*)wsalloc(sizeof(unsigned short) * SLEN * EDIM);
  unsigned short* wihbf = (unsigned short*)wsalloc(sizeof(unsigned short) * 2 * G4 * EDIM);

  k_gather_x<<<(SLEN*EDIM + 255)/256, 256, 0, stream>>>(sent, emb, xbf);
  k_cvt<<<(G4*EDIM + 255)/256, 256, 0, stream>>>(wih_f, wihbf, G4*EDIM);
  k_cvt<<<(G4*EDIM + 255)/256, 256, 0, stream>>>(wih_b, wihbf + (size_t)G4*EDIM, G4*EDIM);
  k_zero<<<(2*SLEN + 255)/256, 256, 0, stream>>>(bar, 2*SLEN);

  dim3 gg(G4/256, SLEN/64, 2);
  k_xproj<<<gg, 256, 0, stream>>>(xbf, wihbf, b_f, b_b, xproj);
  k_lstm<<<64, 256, 0, stream>>>(whh_f, whh_b, h0, c0, xproj, hbuf, hs, bar);
  k_feats<<<SLEN, 384, 0, stream>>>(hs, w_out, b_out, feats);
  k_crf<<<1, 256, 0, stream>>>(feats, gold, trans, out);
}
